// MMGCRN_14096082665704
// MI455X (gfx1250) — matrix operand order repaired and checked by reading, not compiled, not run on hardware
//
#include <hip/hip_runtime.h>
#include <hip/hip_bf16.h>

#define DEV __device__ __forceinline__

typedef __bf16 bf16_t;
typedef __attribute__((ext_vector_type(16))) __bf16 bf16x16;
typedef __attribute__((ext_vector_type(8)))  __bf16 bf16x8;
typedef __attribute__((ext_vector_type(8)))  float  f32x8;

// ---- problem dims (from reference) ----
constexpr int B_   = 64;
constexpr int T_   = 12;
constexpr int N_   = 1024;
constexpr int HOR_ = 12;
constexpr int RNN_ = 64;
constexpr int MEMN_= 10;
constexpr int MEMD_= 32;
constexpr int DEC_ = 96;   // RNN + MEM_D
constexpr int CE_  = 65;   // D_IN + RNN
constexpr int CD_  = 98;   // D_OUT + YCOV + DEC
constexpr int GE_  = 128;  // 2*RNN
constexpr int GD_  = 192;  // 2*DEC
constexpr int NCE_ = B_ * CE_;  // 4160 columns, multiple of 64
constexpr int NCD_ = B_ * CD_;  // 6272 columns, multiple of 64

DEV bf16_t to_bf16(float f) {
  union { unsigned short s; bf16_t b; } u;
  unsigned x = __float_as_uint(f);
  x += 0x7fffu + ((x >> 16) & 1u);   // round-to-nearest-even
  u.s = (unsigned short)(x >> 16);
  return u.b;
}

// ---------------------------------------------------------------------------
// C[M x Ncols] (f32) = A[M x K](bf16,lda) * B, with B supplied TRANSPOSED:
// BT[Ncols x K](bf16,ldbt). Block = 8 waves = 128 rows x 64 cols of C; each
// wave owns 16 rows x 64 cols (4 v8f accumulators). The 64x32 B tile is
// staged into LDS with global_load_async_to_lds_b128 (double-buffered,
// ASYNCcnt-synchronized) and shared by all 8 waves; A streams directly from
// global (each wave owns distinct rows). All dims multiples of 64/32.
// ---------------------------------------------------------------------------
#define SHUF16(x, y) __builtin_shufflevector((x), (y), 0,1,2,3,4,5,6,7,8,9,10,11,12,13,14,15)

constexpr int BROW_ = 40;  // LDS row stride (elements): 80B, 16B-aligned, bank-spread

__global__ __launch_bounds__(256) void wmma_gemm_bf16_kernel(
    const bf16_t* __restrict__ A, const bf16_t* __restrict__ BT,
    float* __restrict__ C, int K, int lda, int ldbt, int ldc)
{
  __shared__ __align__(16) bf16_t Bs[2][64 * BROW_];

  const int tid  = threadIdx.x;
  const int lane = tid & 31;
  const int wave = tid >> 5;
  const int lo   = lane & 15;
  const int hi   = lane >> 4;
  const int row0 = blockIdx.y * 128 + wave * 16;
  const int col0 = blockIdx.x * 64;

  // async staging: thread -> (B-tile row r, 16-byte chunk q)
  const int r = tid >> 2;
  const int q = tid & 3;
  const bf16_t* gsrc = BT + (size_t)(col0 + r) * ldbt + q * 8;
  const unsigned ldsStride = (unsigned)(64 * BROW_ * 2);
  const unsigned lds0 = (unsigned)(uintptr_t)(&Bs[0][0]) +
                        (unsigned)((r * BROW_ + q * 8) * 2);

  const bf16_t* ar = A + (size_t)(row0 + lo) * lda;

  f32x8 c0 = {}, c1 = {}, c2 = {}, c3 = {};

  // preload k=0 tile into buffer 0
  asm volatile("global_load_async_to_lds_b128 %0, %1, off"
               :: "v"(lds0), "v"(gsrc) : "memory");

  for (int k0 = 0; k0 < K; k0 += 32) {
    const int cbuf = (k0 >> 5) & 1;
    const bool more = (k0 + 32) < K;
    if (more) {   // kick off next tile into the other buffer
      asm volatile("global_load_async_to_lds_b128 %0, %1, off"
                   :: "v"(lds0 + (unsigned)(cbuf ^ 1) * ldsStride),
                      "v"(gsrc + k0 + 32) : "memory");
      asm volatile("s_wait_asynccnt 1" ::: "memory");
    } else {
      asm volatile("s_wait_asynccnt 0" ::: "memory");
    }
    __syncthreads();   // current tile visible from all waves

    // A fragment: 16-bit 16x32 layout, lane group hi holds k 8hi..8hi+7 and 16+8hi..16+8hi+7
    bf16x16 av = SHUF16(*(const bf16x8*)(ar + k0 + hi * 8),
                        *(const bf16x8*)(ar + k0 + 16 + hi * 8));
    const bf16_t* bs = &Bs[cbuf][hi * 8];
#define BFRAG(tc) SHUF16(*(const bf16x8*)(bs + (tc * 16 + lo) * BROW_), \
                         *(const bf16x8*)(bs + (tc * 16 + lo) * BROW_ + 16))
    bf16x16 bv0 = BFRAG(0);
    bf16x16 bv1 = BFRAG(1);
    bf16x16 bv2 = BFRAG(2);
    bf16x16 bv3 = BFRAG(3);
#undef BFRAG
    c0 = __builtin_amdgcn_wmma_f32_16x16x32_bf16(false, av, false, bv0, (short)0, c0, false, false);
    c1 = __builtin_amdgcn_wmma_f32_16x16x32_bf16(false, av, false, bv1, (short)0, c1, false, false);
    c2 = __builtin_amdgcn_wmma_f32_16x16x32_bf16(false, av, false, bv2, (short)0, c2, false, false);
    c3 = __builtin_amdgcn_wmma_f32_16x16x32_bf16(false, av, false, bv3, (short)0, c3, false, false);

    __syncthreads();   // all waves done reading before this buffer is restaged
  }

#pragma unroll
  for (int j = 0; j < 8; ++j) {   // C/D layout: VGPR j -> M = j + 8*laneGroup
    float* cr = C + (size_t)(row0 + hi * 8 + j) * ldc + col0 + lo;
    cr[0]  = c0[j];
    cr[16] = c1[j];
    cr[32] = c2[j];
    cr[48] = c3[j];
  }
}

// ---------------------------------------------------------------------------
// helpers
// ---------------------------------------------------------------------------
__global__ __launch_bounds__(256) void zero_f32_kernel(float* p, int n) {
  int i = blockIdx.x * 256 + threadIdx.x;
  if (i < n) p[i] = 0.f;
}

__global__ __launch_bounds__(256) void f32_to_bf16_kernel(const float* __restrict__ in,
                                                          bf16_t* __restrict__ out, int n) {
  int i = blockIdx.x * 256 + threadIdx.x;
  if (i < n) out[i] = to_bf16(in[i]);
}

__global__ __launch_bounds__(256) void transpose_f32_to_bf16_kernel(
    const float* __restrict__ in, bf16_t* __restrict__ out, int rows, int cols) {
  int i = blockIdx.x * 256 + threadIdx.x;
  if (i < rows * cols) {
    int r = i / cols, c = i % cols;
    out[(size_t)c * rows + r] = to_bf16(in[i]);
  }
}

// Aemb[1024 x 32] bf16, zero-padded from node_embeddings[1024 x 8]
__global__ __launch_bounds__(256) void build_aemb_kernel(const float* __restrict__ emb,
                                                         bf16_t* __restrict__ out) {
  int i = blockIdx.x * 256 + threadIdx.x;   // N_*32
  int r = i >> 5, c = i & 31;
  out[i] = to_bf16(c < 8 ? emb[r * 8 + c] : 0.f);
}

// softmax(relu(row)) over `cols` — one block per row
__global__ __launch_bounds__(256) void relu_rowsoftmax_kernel(
    const float* __restrict__ in, float* __restrict__ out, int cols) {
  __shared__ float red[256];
  const int row = blockIdx.x;
  const float* r = in + (size_t)row * cols;
  float* o = out + (size_t)row * cols;
  float mx = 0.f;                                   // relu floor is 0
  for (int c = threadIdx.x; c < cols; c += 256) mx = fmaxf(mx, fmaxf(r[c], 0.f));
  red[threadIdx.x] = mx; __syncthreads();
  for (int s = 128; s > 0; s >>= 1) { if (threadIdx.x < s) red[threadIdx.x] = fmaxf(red[threadIdx.x], red[threadIdx.x + s]); __syncthreads(); }
  mx = red[0]; __syncthreads();
  float sm = 0.f;
  for (int c = threadIdx.x; c < cols; c += 256) { float v = __expf(fmaxf(r[c], 0.f) - mx); o[c] = v; sm += v; }
  red[threadIdx.x] = sm; __syncthreads();
  for (int s = 128; s > 0; s >>= 1) { if (threadIdx.x < s) red[threadIdx.x] += red[threadIdx.x + s]; __syncthreads(); }
  float inv = 1.f / red[0];
  for (int c = threadIdx.x; c < cols; c += 256) o[c] *= inv;
}

// sup2 = bf16(2*S@S - I)
__global__ __launch_bounds__(256) void cheb2_finish_kernel(const float* __restrict__ SS,
                                                           bf16_t* __restrict__ sup2) {
  int i = blockIdx.x * 256 + threadIdx.x;   // N_*N_
  int r = i >> 10, c = i & (N_ - 1);
  sup2[i] = to_bf16(2.f * SS[i] - (r == c ? 1.f : 0.f));
}

// ---------------------------------------------------------------------------
// encoder step kernels
// ---------------------------------------------------------------------------
// XcatT[col = b*CE + c][node] = (c==0 ? x[b,t,n] : h[b,n,c-1])
__global__ __launch_bounds__(256) void build_enc_x_kernel(
    const float* __restrict__ x, int t, const float* __restrict__ h, bf16_t* __restrict__ X) {
  int i = blockIdx.x * 256 + threadIdx.x;   // B_*CE_*N_
  int col = i >> 10, n = i & (N_ - 1);
  int b = col / CE_, c = col % CE_;
  float v = (c == 0) ? x[((size_t)b * T_ + t) * N_ + n]
                     : h[((size_t)b * N_ + n) * RNN_ + (c - 1)];
  X[(size_t)col * N_ + n] = to_bf16(v);
}

// cand: c==0 -> x ; c>=1 -> z*h  (z = zr[...,c-1], already sigmoided)
__global__ __launch_bounds__(256) void build_enc_cand_kernel(
    const float* __restrict__ x, int t, const float* __restrict__ h,
    const float* __restrict__ zr, bf16_t* __restrict__ X) {
  int i = blockIdx.x * 256 + threadIdx.x;
  int col = i >> 10, n = i & (N_ - 1);
  int b = col / CE_, c = col % CE_;
  size_t bn = (size_t)b * N_ + n;
  float v;
  if (c == 0) v = x[((size_t)b * T_ + t) * N_ + n];
  else        v = zr[bn * GE_ + (c - 1)] * h[bn * RNN_ + (c - 1)];
  X[(size_t)col * N_ + n] = to_bf16(v);
}

// zr[b,n,j] = sigmoid( [xt|xg1|xg2] . gW[:,j] + gb[j] ),  j < GE_
__global__ __launch_bounds__(256) void enc_gate_kernel(
    const float* __restrict__ x, int t, const float* __restrict__ h,
    const float* __restrict__ xg1, const float* __restrict__ xg2,
    const float* __restrict__ gW, const float* __restrict__ gb, float* __restrict__ zr) {
  size_t i = (size_t)blockIdx.x * 256 + threadIdx.x;   // B_*N_*GE_
  int j = i % GE_; size_t bn = i / GE_;
  int n = bn & (N_ - 1), b = bn >> 10;
  float acc = gb[j];
  acc += x[((size_t)b * T_ + t) * N_ + n] * gW[0 * GE_ + j];
  const float* hp = h + bn * RNN_;
  for (int c = 1; c < CE_; ++c) acc += hp[c - 1] * gW[c * GE_ + j];
  const float* g1 = xg1 + (size_t)n * NCE_ + b * CE_;
  const float* g2 = xg2 + (size_t)n * NCE_ + b * CE_;
  for (int c = 0; c < CE_; ++c) acc += g1[c] * gW[(CE_ + c) * GE_ + j];
  for (int c = 0; c < CE_; ++c) acc += g2[c] * gW[(2 * CE_ + c) * GE_ + j];
  zr[i] = 1.f / (1.f + __expf(-acc));
}

// hc = tanh([cand|xg1|xg2].uW + ub) ; hnew = r*h + (1-r)*hc ,  j < RNN_
__global__ __launch_bounds__(256) void enc_update_kernel(
    const float* __restrict__ x, int t, const float* __restrict__ h,
    const float* __restrict__ zr, const float* __restrict__ xg1, const float* __restrict__ xg2,
    const float* __restrict__ uW, const float* __restrict__ ub, float* __restrict__ hnew) {
  size_t i = (size_t)blockIdx.x * 256 + threadIdx.x;   // B_*N_*RNN_
  int j = i % RNN_; size_t bn = i / RNN_;
  int n = bn & (N_ - 1), b = bn >> 10;
  float acc = ub[j];
  acc += x[((size_t)b * T_ + t) * N_ + n] * uW[0 * RNN_ + j];
  const float* hp = h + bn * RNN_;
  const float* zp = zr + bn * GE_;
  for (int c = 1; c < CE_; ++c) acc += zp[c - 1] * hp[c - 1] * uW[c * RNN_ + j];
  const float* g1 = xg1 + (size_t)n * NCE_ + b * CE_;
  const float* g2 = xg2 + (size_t)n * NCE_ + b * CE_;
  for (int c = 0; c < CE_; ++c) acc += g1[c] * uW[(CE_ + c) * RNN_ + j];
  for (int c = 0; c < CE_; ++c) acc += g2[c] * uW[(2 * CE_ + c) * RNN_ + j];
  float hc = tanhf(acc);
  float r = zp[RNN_ + j];
  hnew[bn * RNN_ + j] = r * hp[j] + (1.f - r) * hc;
}

// ---------------------------------------------------------------------------
// attention + decoder-init: query/att/proto/W_E/dec_h0, one thread per (b,n)
// ---------------------------------------------------------------------------
__global__ __launch_bounds__(256) void attention_kernel(
    const float* __restrict__ h, const float* __restrict__ Wq, const float* __restrict__ Mem,
    const float* __restrict__ FCE, bf16_t* __restrict__ Agram, float* __restrict__ hdec) {
  size_t bn = (size_t)blockIdx.x * 256 + threadIdx.x;   // B_*N_
  int n = bn & (N_ - 1), b = bn >> 10;
  const float* hp = h + bn * RNN_;
  float q[MEMD_];
#pragma unroll
  for (int d = 0; d < MEMD_; ++d) q[d] = 0.f;
  for (int r = 0; r < RNN_; ++r) {
    float hv = hp[r];
#pragma unroll
    for (int d = 0; d < MEMD_; ++d) q[d] += hv * Wq[r * MEMD_ + d];
  }
  float att[MEMN_]; float mx = -1e30f;
#pragma unroll
  for (int m = 0; m < MEMN_; ++m) {
    float s = 0.f;
#pragma unroll
    for (int d = 0; d < MEMD_; ++d) s += q[d] * Mem[m * MEMD_ + d];
    att[m] = s; mx = fmaxf(mx, s);
  }
  float sm = 0.f;
#pragma unroll
  for (int m = 0; m < MEMN_; ++m) { att[m] = __expf(att[m] - mx); sm += att[m]; }
  float inv = 1.f / sm;
  float p[MEMD_];
#pragma unroll
  for (int d = 0; d < MEMD_; ++d) {
    float s = 0.f;
#pragma unroll
    for (int m = 0; m < MEMN_; ++m) s += att[m] * Mem[m * MEMD_ + d];
    p[d] = s * inv;
  }
  // W_E = proto @ FC_E  -> Agram[n, b*8+e] (for the S_dec gram GEMM)
#pragma unroll
  for (int e = 0; e < 8; ++e) {
    float s = 0.f;
#pragma unroll
    for (int d = 0; d < MEMD_; ++d) s += p[d] * FCE[d * 8 + e];
    Agram[(size_t)n * (B_ * 8) + b * 8 + e] = to_bf16(s);
  }
  // dec_h0 = [h | proto]
  float* hd = hdec + bn * DEC_;
#pragma unroll
  for (int j = 0; j < RNN_; ++j) hd[j] = hp[j];
#pragma unroll
  for (int d = 0; d < MEMD_; ++d) hd[RNN_ + d] = p[d];
}

// ---------------------------------------------------------------------------
// decoder step kernels
// ---------------------------------------------------------------------------
__global__ __launch_bounds__(256) void build_dec_x_kernel(
    const float* __restrict__ go, const float* __restrict__ yc, int t,
    const float* __restrict__ h, bf16_t* __restrict__ X) {
  int i = blockIdx.x * 256 + threadIdx.x;   // B_*CD_*N_
  int col = i >> 10, n = i & (N_ - 1);
  int b = col / CD_, c = col % CD_;
  size_t bn = (size_t)b * N_ + n;
  float v;
  if (c == 0)      v = go[bn];
  else if (c == 1) v = yc[((size_t)b * HOR_ + t) * N_ + n];
  else             v = h[bn * DEC_ + (c - 2)];
  X[(size_t)col * N_ + n] = to_bf16(v);
}

__global__ __launch_bounds__(256) void build_dec_cand_kernel(
    const float* __restrict__ go, const float* __restrict__ yc, int t,
    const float* __restrict__ h, const float* __restrict__ zr, bf16_t* __restrict__ X) {
  int i = blockIdx.x * 256 + threadIdx.x;
  int col = i >> 10, n = i & (N_ - 1);
  int b = col / CD_, c = col % CD_;
  size_t bn = (size_t)b * N_ + n;
  float v;
  if (c == 0)      v = go[bn];
  else if (c == 1) v = yc[((size_t)b * HOR_ + t) * N_ + n];
  else             v = zr[bn * GD_ + (c - 2)] * h[bn * DEC_ + (c - 2)];
  X[(size_t)col * N_ + n] = to_bf16(v);
}

__global__ __launch_bounds__(256) void dec_gate_kernel(
    const float* __restrict__ go, const float* __restrict__ yc, int t,
    const float* __restrict__ h, const float* __restrict__ xg1, const float* __restrict__ xg2,
    const float* __restrict__ gW, const float* __restrict__ gb, float* __restrict__ zr) {
  size_t i = (size_t)blockIdx.x * 256 + threadIdx.x;   // B_*N_*GD_
  int j = i % GD_; size_t bn = i / GD_;
  int n = bn & (N_ - 1), b = bn >> 10;
  float acc = gb[j];
  acc += go[bn] * gW[0 * GD_ + j];
  acc += yc[((size_t)b * HOR_ + t) * N_ + n] * gW[1 * GD_ + j];
  const float* hp = h + bn * DEC_;
  for (int c = 2; c < CD_; ++c) acc += hp[c - 2] * gW[c * GD_ + j];
  const float* g1 = xg1 + (size_t)n * NCD_ + b * CD_;
  const float* g2 = xg2 + (size_t)n * NCD_ + b * CD_;
  for (int c = 0; c < CD_; ++c) acc += g1[c] * gW[(CD_ + c) * GD_ + j];
  for (int c = 0; c < CD_; ++c) acc += g2[c] * gW[(2 * CD_ + c) * GD_ + j];
  zr[i] = 1.f / (1.f + __expf(-acc));
}

__global__ __launch_bounds__(256) void dec_update_kernel(
    const float* __restrict__ go, const float* __restrict__ yc, int t,
    const float* __restrict__ h, const float* __restrict__ zr,
    const float* __restrict__ xg1, const float* __restrict__ xg2,
    const float* __restrict__ uW, const float* __restrict__ ub, float* __restrict__ hnew) {
  size_t i = (size_t)blockIdx.x * 256 + threadIdx.x;   // B_*N_*DEC_
  int j = i % DEC_; size_t bn = i / DEC_;
  int n = bn & (N_ - 1), b = bn >> 10;
  float acc = ub[j];
  acc += go[bn] * uW[0 * DEC_ + j];
  acc += yc[((size_t)b * HOR_ + t) * N_ + n] * uW[1 * DEC_ + j];
  const float* hp = h + bn * DEC_;
  const float* zp = zr + bn * GD_;
  for (int c = 2; c < CD_; ++c) acc += zp[c - 2] * hp[c - 2] * uW[c * DEC_ + j];
  const float* g1 = xg1 + (size_t)n * NCD_ + b * CD_;
  const float* g2 = xg2 + (size_t)n * NCD_ + b * CD_;
  for (int c = 0; c < CD_; ++c) acc += g1[c] * uW[(CD_ + c) * DEC_ + j];
  for (int c = 0; c < CD_; ++c) acc += g2[c] * uW[(2 * CD_ + c) * DEC_ + j];
  float hc = tanhf(acc);
  float r = zp[DEC_ + j];
  hnew[bn * DEC_ + j] = r * hp[j] + (1.f - r) * hc;
}

// go_new = h.proj_W + proj_b ; also scatter into output (B,HOR,N)
__global__ __launch_bounds__(256) void dec_proj_kernel(
    const float* __restrict__ h, const float* __restrict__ pW, const float* __restrict__ pb,
    float* __restrict__ go, float* __restrict__ out, int t) {
  size_t bn = (size_t)blockIdx.x * 256 + threadIdx.x;   // B_*N_
  int n = bn & (N_ - 1), b = bn >> 10;
  const float* hp = h + bn * DEC_;
  float acc = pb[0];
#pragma unroll
  for (int j = 0; j < DEC_; ++j) acc += hp[j] * pW[j];
  go[bn] = acc;
  out[((size_t)b * HOR_ + t) * N_ + n] = acc;
}

// ---------------------------------------------------------------------------
extern "C" void kernel_launch(void* const* d_in, const int* in_sizes, int n_in,
                              void* d_out, int out_size, void* d_ws, size_t ws_size,
                              hipStream_t stream) {
  (void)in_sizes; (void)n_in; (void)out_size; (void)ws_size;
  const float* x    = (const float*)d_in[0];
  const float* ycov = (const float*)d_in[1];
  const float* emb  = (const float*)d_in[2];
  const float* egW  = (const float*)d_in[3];
  const float* egb  = (const float*)d_in[4];
  const float* euW  = (const float*)d_in[5];
  const float* eub  = (const float*)d_in[6];
  const float* Mem  = (const float*)d_in[7];
  const float* Wq   = (const float*)d_in[8];
  const float* FCE  = (const float*)d_in[9];
  const float* dgW  = (const float*)d_in[10];
  const float* dgb  = (const float*)d_in[11];
  const float* duW  = (const float*)d_in[12];
  const float* dub  = (const float*)d_in[13];
  const float* pW   = (const float*)d_in[14];
  const float* pb   = (const float*)d_in[15];
  float* out = (float*)d_out;

  char* w = (char*)d_ws;
  auto alloc = [&](size_t bytes) { char* p = w; w += (bytes + 255) & ~(size_t)255; return p; };
  const size_t NN = (size_t)N_ * N_;
  float*  Sg     = (float*)alloc(NN * 4);                 // gram / S@S f32 scratch
  float*  Sden   = (float*)alloc(NN * 4);                 // dense S f32 (enc, then dec)
  bf16_t* S_bf   = (bf16_t*)alloc(NN * 2);
  bf16_t* ST_bf  = (bf16_t*)alloc(NN * 2);
  bf16_t* S2_bf  = (bf16_t*)alloc(NN * 2);                // 2*S@S - I
  bf16_t* Aemb   = (bf16_t*)alloc((size_t)N_ * 32 * 2);
  bf16_t* Agram  = (bf16_t*)alloc((size_t)N_ * B_ * 8 * 2);
  bf16_t* X1     = (bf16_t*)alloc((size_t)NCD_ * N_ * 2); // transposed batched features
  float*  xg1    = (float*)alloc((size_t)N_ * NCD_ * 4);
  float*  xg2    = (float*)alloc((size_t)N_ * NCD_ * 4);
  float*  zr     = (float*)alloc((size_t)B_ * N_ * GD_ * 4);
  float*  hA     = (float*)alloc((size_t)B_ * N_ * DEC_ * 4);
  float*  hB     = (float*)alloc((size_t)B_ * N_ * DEC_ * 4);
  float*  go     = (float*)alloc((size_t)B_ * N_ * 4);

  auto gemm = [&](const bf16_t* A, const bf16_t* BT, float* C, int K, int Ncols,
                  int lda, int ldbt, int ldc) {
    wmma_gemm_bf16_kernel<<<dim3(Ncols / 64, N_ / 128), 256, 0, stream>>>(A, BT, C, K, lda, ldbt, ldc);
  };
  auto make_supports = [&](float* Sf32) {  // Sf32 -> S_bf, ST_bf, S2_bf
    f32_to_bf16_kernel<<<NN / 256, 256, 0, stream>>>(Sf32, S_bf, (int)NN);
    transpose_f32_to_bf16_kernel<<<NN / 256, 256, 0, stream>>>(Sf32, ST_bf, N_, N_);
    gemm(S_bf, ST_bf, Sg, N_, N_, N_, N_, N_);           // Sg = S @ S
    cheb2_finish_kernel<<<NN / 256, 256, 0, stream>>>(Sg, S2_bf);
  };

  // ---- encoder supports: S_enc = softmax(relu(emb @ emb^T)) ----
  build_aemb_kernel<<<(N_ * 32) / 256, 256, 0, stream>>>(emb, Aemb);
  gemm(Aemb, Aemb, Sg, 32, N_, 32, 32, N_);
  relu_rowsoftmax_kernel<<<N_, 256, 0, stream>>>(Sg, Sden, N_);
  make_supports(Sden);

  // ---- encoder GRU ----
  zero_f32_kernel<<<(B_ * N_ * RNN_) / 256, 256, 0, stream>>>(hA, B_ * N_ * RNN_);
  float* hc = hA; float* hn = hB;
  for (int t = 0; t < T_; ++t) {
    build_enc_x_kernel<<<(B_ * CE_ * N_) / 256, 256, 0, stream>>>(x, t, hc, X1);
    gemm(S_bf, X1, xg1, N_, NCE_, N_, N_, NCE_);
    gemm(S2_bf, X1, xg2, N_, NCE_, N_, N_, NCE_);
    enc_gate_kernel<<<(B_ * N_ * GE_) / 256, 256, 0, stream>>>(x, t, hc, xg1, xg2, egW, egb, zr);
    build_enc_cand_kernel<<<(B_ * CE_ * N_) / 256, 256, 0, stream>>>(x, t, hc, zr, X1);
    gemm(S_bf, X1, xg1, N_, NCE_, N_, N_, NCE_);
    gemm(S2_bf, X1, xg2, N_, NCE_, N_, N_, NCE_);
    enc_update_kernel<<<(B_ * N_ * RNN_) / 256, 256, 0, stream>>>(x, t, hc, zr, xg1, xg2, euW, eub, hn);
    float* tmp = hc; hc = hn; hn = tmp;
  }
  // hc == hA after 12 swaps (even count)

  // ---- attention / memory: proto, W_E gram input, dec_h0 into hn(=hB) ----
  attention_kernel<<<(B_ * N_) / 256, 256, 0, stream>>>(hc, Wq, Mem, FCE, Agram, hn);

  // ---- decoder supports: S_dec = softmax(relu(einsum(bnc,bmc->nm))) ----
  gemm(Agram, Agram, Sg, B_ * 8, N_, B_ * 8, B_ * 8, N_);
  relu_rowsoftmax_kernel<<<N_, 256, 0, stream>>>(Sg, Sden, N_);
  make_supports(Sden);

  // ---- decoder GRU ----
  zero_f32_kernel<<<(B_ * N_) / 256, 256, 0, stream>>>(go, B_ * N_);
  hc = hn; hn = (hc == hA) ? hB : hA;
  for (int t = 0; t < HOR_; ++t) {
    build_dec_x_kernel<<<(B_ * CD_ * N_) / 256, 256, 0, stream>>>(go, ycov, t, hc, X1);
    gemm(S_bf, X1, xg1, N_, NCD_, N_, N_, NCD_);
    gemm(S2_bf, X1, xg2, N_, NCD_, N_, N_, NCD_);
    dec_gate_kernel<<<(B_ * N_ * GD_) / 256, 256, 0, stream>>>(go, ycov, t, hc, xg1, xg2, dgW, dgb, zr);
    build_dec_cand_kernel<<<(B_ * CD_ * N_) / 256, 256, 0, stream>>>(go, ycov, t, hc, zr, X1);
    gemm(S_bf, X1, xg1, N_, NCD_, N_, N_, NCD_);
    gemm(S2_bf, X1, xg2, N_, NCD_, N_, N_, NCD_);
    dec_update_kernel<<<(B_ * N_ * DEC_) / 256, 256, 0, stream>>>(go, ycov, t, hc, zr, xg1, xg2, duW, dub, hn);
    dec_proj_kernel<<<(B_ * N_) / 256, 256, 0, stream>>>(hn, pW, pb, go, out, t);
    float* tmp = hc; hc = hn; hn = tmp;
  }
}
